// DenseMoE_40072044871690
// MI455X (gfx1250) — matrix operand order repaired and checked by reading, not compiled, not run on hardware
//
#include <hip/hip_runtime.h>
#include <stdint.h>

// Problem constants (from reference setup_inputs)
#define TT 2048
#define DD 2048
#define FF 7168
#define EE 8

// GEMM tiling
#define BM 128
#define BN 64     // gemm1 N tile
#define BN2 128   // gemm2 N tile
#define KB 64
#define LDK (KB + 8)   // +16B pad per row

typedef __attribute__((ext_vector_type(16))) __bf16        v16bf;
typedef __attribute__((ext_vector_type(8)))  float         v8f;
typedef __attribute__((ext_vector_type(4)))  unsigned int  u32x4;
typedef __attribute__((ext_vector_type(4)))  float         f32x4;

// Async global->LDS (CDNA5). Guarded: fall back to sync copy if builtin absent.
#if defined(__has_builtin)
#if __has_builtin(__builtin_amdgcn_global_load_async_to_lds_b128)
#define HAVE_ASYNC_LDS 1
#endif
#endif

// Exact operand types per hipcc diagnostic: param0 is
// 'int __attribute__((vector_size(16))) __device__ *'  (addrspace(1) int4*)
typedef int v4i __attribute__((vector_size(16)));
typedef __attribute__((address_space(1))) v4i as1_v4i;
typedef __attribute__((address_space(3))) v4i as3_v4i;

__device__ __forceinline__ void wait_async0() {
#if defined(__has_builtin)
#if __has_builtin(__builtin_amdgcn_s_wait_asynccnt)
  __builtin_amdgcn_s_wait_asynccnt(0);
#else
  asm volatile("s_wait_asynccnt 0x0" ::: "memory");
#endif
#else
  asm volatile("s_wait_asynccnt 0x0" ::: "memory");
#endif
}

// copy 16B global(bf16) -> LDS, async if available
__device__ __forceinline__ void cp16(const unsigned short* g, unsigned short* l) {
#if HAVE_ASYNC_LDS
  __builtin_amdgcn_global_load_async_to_lds_b128(
      (as1_v4i*)g, (as3_v4i*)l, 0, 0);
#else
  *(u32x4*)l = *(const u32x4*)g;
#endif
}

// fp32 -> bf16, round-to-nearest-even
__device__ __forceinline__ unsigned short f2bf(float f) {
  unsigned int u = __builtin_bit_cast(unsigned int, f);
  u += 0x7FFFu + ((u >> 16) & 1u);
  return (unsigned short)(u >> 16);
}

// A fragment (16x32 bf16): lane<16 -> row M=lane, K {0..7, 16..23}
//                          lane>=16 -> row M=lane-16, K {8..15, 24..31}
__device__ __forceinline__ v16bf frag_a(const unsigned short* p) {
  union { v16bf v; u32x4 q[2]; } u;
  u.q[0] = *(const u32x4*)(p);
  u.q[1] = *(const u32x4*)(p + 16);
  return u.v;
}

// B fragment (32x16 bf16): lane = N column (lane&15);
// lanes 0..15 hold K {0..7, 16..23}, lanes 16..31 hold K {8..15, 24..31}
__device__ __forceinline__ v16bf frag_b(const unsigned short* p) {
  union { v16bf v; u32x4 q[2]; } u;
  u.q[0] = *(const u32x4*)(p);
  u.q[1] = *(const u32x4*)(p + 16);
  return u.v;
}

// ---------------------------------------------------------------------------
// Kernel 1: gate logits -> softmax -> top-2 -> renormalized weights (T,E)
//           + convert x row to bf16 (xb)
// ---------------------------------------------------------------------------
__global__ __launch_bounds__(256) void moe_gate(
    const float* __restrict__ x, const float* __restrict__ gw,
    unsigned short* __restrict__ xb, float* __restrict__ wt) {
  __shared__ float red[EE * 256];
  const int t = blockIdx.x;
  const int tid = threadIdx.x;

  float p[EE];
#pragma unroll
  for (int e = 0; e < EE; ++e) p[e] = 0.f;

  for (int d = tid; d < DD; d += 256) {
    float xv = x[(size_t)t * DD + d];
    xb[(size_t)t * DD + d] = f2bf(xv);
#pragma unroll
    for (int e = 0; e < EE; ++e) p[e] += xv * gw[e * DD + d];
  }
#pragma unroll
  for (int e = 0; e < EE; ++e) red[e * 256 + tid] = p[e];
  __syncthreads();
  for (int off = 128; off >= 1; off >>= 1) {
    if (tid < off) {
#pragma unroll
      for (int e = 0; e < EE; ++e)
        red[e * 256 + tid] += red[e * 256 + tid + off];
    }
    __syncthreads();
  }
  if (tid == 0) {
    float l[EE];
    float m = -1e30f;
#pragma unroll
    for (int e = 0; e < EE; ++e) { l[e] = red[e * 256]; m = fmaxf(m, l[e]); }
    float ex[EE];
#pragma unroll
    for (int e = 0; e < EE; ++e) ex[e] = __expf(l[e] - m);
    int i1 = 0;
#pragma unroll
    for (int e = 1; e < EE; ++e) if (ex[e] > ex[i1]) i1 = e;
    int i2 = (i1 == 0) ? 1 : 0;
#pragma unroll
    for (int e = 0; e < EE; ++e) if (e != i1 && ex[e] > ex[i2]) i2 = e;
    float s = ex[i1] + ex[i2];
#pragma unroll
    for (int e = 0; e < EE; ++e)
      wt[(size_t)t * EE + e] = (e == i1 || e == i2) ? (ex[e] / s) : 0.f;
  }
}

// ---------------------------------------------------------------------------
// Kernel 2: per expert: h[t,f] = silu(x.w1^T) * (x.w3^T) * wt[t,e]  (bf16 out)
//   M=T, N=F, K=D. A = xb (bf16, async->LDS); B = w1/w3 (fp32 -> bf16 in LDS).
// ---------------------------------------------------------------------------
__global__ __launch_bounds__(256) void moe_gemm1(
    const unsigned short* __restrict__ xb, const float* __restrict__ w1e,
    const float* __restrict__ w3e, const float* __restrict__ wt,
    unsigned short* __restrict__ hb, int e) {
  __shared__ unsigned short lA[BM * LDK];
  __shared__ unsigned short lB1[BN * LDK];
  __shared__ unsigned short lB3[BN * LDK];

  const int tid  = threadIdx.x;
  const int lane = tid & 31;
  const int wave = tid >> 5;
  const int m0 = blockIdx.y * BM;
  const int n0 = blockIdx.x * BN;
  const int wm = (wave >> 1) * 32;  // 4 wave-rows
  const int wn = (wave & 1) * 32;   // 2 wave-cols

  // hoist per-thread chunk pointers out of the K loop
  const unsigned short* pA[4]; int oA[4];
  const float* pB1[4]; const float* pB3[4]; int oB[4];
#pragma unroll
  for (int i = 0; i < 4; ++i) {
    int c = tid + i * 256;
    { int r = c >> 3, kc = (c & 7) << 3;
      pA[i] = xb + (size_t)(m0 + r) * DD + kc;
      oA[i] = r * LDK + kc; }
    { int r = c >> 4, kc = (c & 15) << 2;
      pB1[i] = w1e + (size_t)(n0 + r) * DD + kc;
      pB3[i] = w3e + (size_t)(n0 + r) * DD + kc;
      oB[i] = r * LDK + kc; }
  }

  v8f gacc[2][2] = {};
  v8f uacc[2][2] = {};

  for (int k0 = 0; k0 < DD; k0 += KB) {
    // A tile: async DMA global->LDS (ASYNCcnt), overlaps with B conversion
#pragma unroll
    for (int i = 0; i < 4; ++i) cp16(pA[i] + k0, lA + oA[i]);
    // B tiles: fp32 load -> bf16 convert -> LDS
#pragma unroll
    for (int i = 0; i < 4; ++i) {
      f32x4 a = *(const f32x4*)(pB1[i] + k0);
      f32x4 b = *(const f32x4*)(pB3[i] + k0);
      unsigned short* q1 = lB1 + oB[i];
      unsigned short* q3 = lB3 + oB[i];
#pragma unroll
      for (int j = 0; j < 4; ++j) { q1[j] = f2bf(a[j]); q3[j] = f2bf(b[j]); }
    }
    if (k0 + KB < DD) {
      __builtin_prefetch(pB1[0] + k0 + KB, 0, 0);
      __builtin_prefetch(pB3[0] + k0 + KB, 0, 0);
    }
#if HAVE_ASYNC_LDS
    wait_async0();
#endif
    __syncthreads();

#pragma unroll
    for (int kk = 0; kk < KB; kk += 32) {
      const int ra  = wm + (lane & 15);
      const int rb  = wn + (lane & 15);
      const int koA = kk + ((lane >> 4) << 3);
      const int koB = kk + ((lane >> 4) << 3);
      v16bf af[2], b1f[2], b3f[2];
      af[0]  = frag_a(&lA [ ra        * LDK + koA]);
      af[1]  = frag_a(&lA [(ra + 16)  * LDK + koA]);
      b1f[0] = frag_b(&lB1[ rb        * LDK + koB]);
      b1f[1] = frag_b(&lB1[(rb + 16)  * LDK + koB]);
      b3f[0] = frag_b(&lB3[ rb        * LDK + koB]);
      b3f[1] = frag_b(&lB3[(rb + 16)  * LDK + koB]);
#pragma unroll
      for (int mt = 0; mt < 2; ++mt)
#pragma unroll
        for (int nt = 0; nt < 2; ++nt) {
          gacc[mt][nt] = __builtin_amdgcn_wmma_f32_16x16x32_bf16(
              false, af[mt], false, b1f[nt], (short)0, gacc[mt][nt], false, false);
          uacc[mt][nt] = __builtin_amdgcn_wmma_f32_16x16x32_bf16(
              false, af[mt], false, b3f[nt], (short)0, uacc[mt][nt], false, false);
        }
    }
    __syncthreads();
  }

  // epilogue: silu(g)*u*wt -> bf16 hb
#pragma unroll
  for (int mt = 0; mt < 2; ++mt)
#pragma unroll
    for (int nt = 0; nt < 2; ++nt) {
      const int colf  = n0 + wn + nt * 16 + (lane & 15);
      const int rbase = m0 + wm + mt * 16 + ((lane >> 4) << 3);
#pragma unroll
      for (int v = 0; v < 8; ++v) {
        int t = rbase + v;
        float g = gacc[mt][nt][v];
        float u = uacc[mt][nt][v];
        float h = (g / (1.f + __expf(-g))) * u * wt[(size_t)t * EE + e];
        hb[(size_t)t * FF + colf] = f2bf(h);
      }
    }
}

// ---------------------------------------------------------------------------
// Kernel 3: per expert: out[t,d] (+)= sum_f h[t,f] * w2[e,d,f]
//   M=T, N=D, K=F. 128x128 block, 8 waves (4x2), wave tile 32x64.
// ---------------------------------------------------------------------------
__global__ __launch_bounds__(256) void moe_gemm2(
    const unsigned short* __restrict__ hb, const float* __restrict__ w2e,
    float* __restrict__ out, int accumulate) {
  __shared__ unsigned short lA[BM * LDK];
  __shared__ unsigned short lB[BN2 * LDK];

  const int tid  = threadIdx.x;
  const int lane = tid & 31;
  const int wave = tid >> 5;
  const int m0 = blockIdx.y * BM;
  const int n0 = blockIdx.x * BN2;
  const int wm = (wave >> 1) * 32;  // 4 wave-rows of 32
  const int wn = (wave & 1) * 64;   // 2 wave-cols of 64

  const unsigned short* pA[4]; int oA[4];
  const float* pB[8]; int oB[8];
#pragma unroll
  for (int i = 0; i < 4; ++i) {
    int c = tid + i * 256;
    int r = c >> 3, kc = (c & 7) << 3;
    pA[i] = hb + (size_t)(m0 + r) * FF + kc;
    oA[i] = r * LDK + kc;
  }
#pragma unroll
  for (int i = 0; i < 8; ++i) {
    int c = tid + i * 256;
    int r = c >> 4, kc = (c & 15) << 2;
    pB[i] = w2e + (size_t)(n0 + r) * FF + kc;
    oB[i] = r * LDK + kc;
  }

  v8f acc[2][4] = {};

  for (int k0 = 0; k0 < FF; k0 += KB) {
#pragma unroll
    for (int i = 0; i < 4; ++i) cp16(pA[i] + k0, lA + oA[i]);
#pragma unroll
    for (int i = 0; i < 8; ++i) {
      f32x4 a = *(const f32x4*)(pB[i] + k0);
      unsigned short* q = lB + oB[i];
#pragma unroll
      for (int j = 0; j < 4; ++j) q[j] = f2bf(a[j]);
    }
    if (k0 + KB < FF) __builtin_prefetch(pB[0] + k0 + KB, 0, 0);
#if HAVE_ASYNC_LDS
    wait_async0();
#endif
    __syncthreads();

#pragma unroll
    for (int kk = 0; kk < KB; kk += 32) {
      const int ra  = wm + (lane & 15);
      const int rb  = wn + (lane & 15);
      const int koA = kk + ((lane >> 4) << 3);
      const int koB = kk + ((lane >> 4) << 3);
      v16bf af[2], bf[4];
      af[0] = frag_a(&lA[ ra       * LDK + koA]);
      af[1] = frag_a(&lA[(ra + 16) * LDK + koA]);
#pragma unroll
      for (int j = 0; j < 4; ++j)
        bf[j] = frag_b(&lB[(rb + 16 * j) * LDK + koB]);
#pragma unroll
      for (int mt = 0; mt < 2; ++mt)
#pragma unroll
        for (int nt = 0; nt < 4; ++nt)
          acc[mt][nt] = __builtin_amdgcn_wmma_f32_16x16x32_bf16(
              false, af[mt], false, bf[nt], (short)0, acc[mt][nt], false, false);
    }
    __syncthreads();
  }

#pragma unroll
  for (int mt = 0; mt < 2; ++mt)
#pragma unroll
    for (int nt = 0; nt < 4; ++nt) {
      const int col   = n0 + wn + nt * 16 + (lane & 15);
      const int rbase = m0 + wm + mt * 16 + ((lane >> 4) << 3);
#pragma unroll
      for (int v = 0; v < 8; ++v) {
        size_t idx = (size_t)(rbase + v) * DD + col;
        float val = acc[mt][nt][v];
        if (accumulate) val += out[idx];
        out[idx] = val;
      }
    }
}

// ---------------------------------------------------------------------------
extern "C" void kernel_launch(void* const* d_in, const int* in_sizes, int n_in,
                              void* d_out, int out_size, void* d_ws, size_t ws_size,
                              hipStream_t stream) {
  (void)in_sizes; (void)n_in; (void)out_size; (void)ws_size;
  const float* x  = (const float*)d_in[0];
  const float* gw = (const float*)d_in[1];
  const float* w1 = (const float*)d_in[2];
  const float* w3 = (const float*)d_in[3];
  const float* w2 = (const float*)d_in[4];
  float* out = (float*)d_out;

  char* ws = (char*)d_ws;
  unsigned short* xb = (unsigned short*)ws;                                   // T*D bf16 (8 MB)
  float*          wt = (float*)(ws + (size_t)TT * DD * 2);                    // T*E f32  (64 KB)
  unsigned short* hb = (unsigned short*)(ws + (size_t)TT * DD * 2
                                            + (size_t)TT * EE * 4);           // T*F bf16 (28.7 MB)

  moe_gate<<<TT, 256, 0, stream>>>(x, gw, xb, wt);
  for (int e = 0; e < EE; ++e) {
    moe_gemm1<<<dim3(FF / BN, TT / BM), 256, 0, stream>>>(
        xb, w1 + (size_t)e * FF * DD, w3 + (size_t)e * FF * DD, wt, hb, e);
    moe_gemm2<<<dim3(DD / BN2, TT / BM), 256, 0, stream>>>(
        hb, w2 + (size_t)e * DD * FF, out, /*accumulate=*/e > 0);
  }
}
